// SwitchTransformersLayerFF_29858612642049
// MI455X (gfx1250) — matrix operand order repaired and checked by reading, not compiled, not run on hardware
//
#include <hip/hip_runtime.h>
#include <stdint.h>

// ---------------- problem constants ----------------
constexpr int kTok = 16384;       // B*S = 8*2048
constexpr int kD   = 768;
constexpr int kF   = 3072;
constexpr int kE   = 8;
constexpr int kD16 = kD / 16;     // 48
constexpr int kD32 = kD / 32;     // 24
constexpr int kF16 = kF / 16;     // 192
constexpr int kF32 = kF / 32;     // 96
constexpr int MT   = 32;          // tokens per tile
constexpr int XROW = 776;         // padded LDS row (shorts) for x  (768+8)
constexpr int HROW = 72;          // padded LDS row (shorts) for h  (64+8)
constexpr int FC   = 64;          // F chunk
constexpr int NCH  = kF / FC;     // 48 chunks

typedef __attribute__((ext_vector_type(16))) __bf16 bf16x16;
typedef __attribute__((ext_vector_type(8)))  float  floatx8;

union Frag { uint4 q[2]; bf16x16 v; };

__device__ __forceinline__ unsigned short f2bf(float f) {
  union { float f; unsigned int u; } v; v.f = f;
  unsigned int u = v.u;
  unsigned int r = (u + 0x7FFFu + ((u >> 16) & 1u)) >> 16;   // RNE
  return (unsigned short)r;
}

__device__ __forceinline__ bf16x16 ldsFrag(const unsigned short* p0, const unsigned short* p1) {
  Frag f;
  f.q[0] = *(const uint4*)p0;   // v16bf slots 0..7
  f.q[1] = *(const uint4*)p1;   // v16bf slots 8..15
  return f.v;
}

__device__ __forceinline__ bf16x16 gFrag(const unsigned short* p) {
  Frag f;
  const uint4* q = (const uint4*)p;
  f.q[0] = q[0];
  f.q[1] = q[1];
  return f.v;
}

// meta layout (ints): [0..7]=counts  [8..15]=fill  [16..24]=offsets  [25..33]=tile_off
__global__ void init_meta_kernel(int* meta) {
  if (threadIdx.x < 16) meta[threadIdx.x] = 0;
}

// fp32 [E][K][N] row-major  ->  bf16 WMMA-B-fragment-major tiles.
// tile t = (e*(N/16)+nt)*(K/32)+kt ; within tile: lane l (0..31) holds 16 shorts,
// slot j <-> element B[kt*32 + (l>=16?16:0) + j][nt*16 + (l&15)]
__global__ __launch_bounds__(256) void pack_kernel(const float* __restrict__ src,
                                                   unsigned short* __restrict__ dst,
                                                   int K, int Ndim) {
  long gid = (long)blockIdx.x * 256 + threadIdx.x;
  int within = (int)(gid & 511);
  long t = gid >> 9;
  int l = within >> 4, j = within & 15;
  int ktiles = K >> 5, ntiles = Ndim >> 4;
  int kt = (int)(t % ktiles);
  long rest = t / ktiles;
  int nt = (int)(rest % ntiles);
  int e  = (int)(rest / ntiles);
  int k = kt * 32 + ((l >= 16) ? 16 : 0) + j;
  int n = nt * 16 + (l & 15);
  float v = src[((long)e * K + k) * Ndim + n];
  dst[gid] = f2bf(v);
}

// One wave32 per token: RMSNorm, bf16 xn, router logits, softmax/argmax/top-prob.
__global__ __launch_bounds__(256) void router_kernel(const float* __restrict__ hs,
                                                     const float* __restrict__ lnw,
                                                     const float* __restrict__ rw,
                                                     unsigned short* __restrict__ xn,
                                                     float* __restrict__ probs,
                                                     int* __restrict__ expert_of,
                                                     int* __restrict__ meta,
                                                     float* __restrict__ logits_out,
                                                     float* __restrict__ idx_out) {
  int wid = threadIdx.x >> 5, lane = threadIdx.x & 31;
  int token = blockIdx.x * 8 + wid;
  const float* row = hs + (long)token * kD;

  float vx[24];
  float ss = 0.f;
#pragma unroll
  for (int it = 0; it < 24; ++it) { float v = row[it * 32 + lane]; vx[it] = v; ss += v * v; }
#pragma unroll
  for (int off = 16; off >= 1; off >>= 1) ss += __shfl_xor(ss, off, 32);
  float rs = rsqrtf(ss * (1.0f / kD) + 1e-6f);

  float lg[8];
#pragma unroll
  for (int e2 = 0; e2 < 8; ++e2) lg[e2] = 0.f;

  unsigned short* xrow = xn + (long)token * kD;
#pragma unroll
  for (int it = 0; it < 24; ++it) {
    int d = it * 32 + lane;
    float xv = vx[it] * rs * lnw[d];
    xrow[d] = f2bf(xv);
    const float4* r4 = (const float4*)(rw + d * 8);
    float4 ra = r4[0], rb = r4[1];
    lg[0] += xv * ra.x; lg[1] += xv * ra.y; lg[2] += xv * ra.z; lg[3] += xv * ra.w;
    lg[4] += xv * rb.x; lg[5] += xv * rb.y; lg[6] += xv * rb.z; lg[7] += xv * rb.w;
  }
#pragma unroll
  for (int e2 = 0; e2 < 8; ++e2) {
#pragma unroll
    for (int off = 16; off >= 1; off >>= 1) lg[e2] += __shfl_xor(lg[e2], off, 32);
  }
  if (lane == 0) {
    float m = lg[0]; int a = 0;
#pragma unroll
    for (int e2 = 1; e2 < 8; ++e2) if (lg[e2] > m) { m = lg[e2]; a = e2; }   // first max
    float s = 0.f;
#pragma unroll
    for (int e2 = 0; e2 < 8; ++e2) s += __expf(lg[e2] - m);
    float p = 1.0f / s;                // softmax value at argmax
#pragma unroll
    for (int e2 = 0; e2 < 8; ++e2) logits_out[(long)token * 8 + e2] = lg[e2];
    idx_out[token]   = (float)a;
    probs[token]     = p;
    expert_of[token] = a;
    atomicAdd(&meta[a], 1);
  }
}

__global__ void scan_kernel(int* meta) {
  if (threadIdx.x == 0 && blockIdx.x == 0) {
    int* counts = meta; int* fill = meta + 8; int* off = meta + 16; int* toff = meta + 25;
    off[0] = 0; toff[0] = 0;
    for (int e = 0; e < 8; ++e) {
      off[e + 1]  = off[e] + counts[e];
      toff[e + 1] = toff[e] + ((counts[e] + MT - 1) / MT);
      fill[e]     = off[e];
    }
  }
}

__global__ __launch_bounds__(256) void scatter_kernel(const int* __restrict__ expert_of,
                                                      int* meta, int* __restrict__ bucket) {
  int t = blockIdx.x * 256 + threadIdx.x;
  int e = expert_of[t];
  int pos = atomicAdd(&meta[8 + e], 1);
  bucket[pos] = t;
}

// Core fused expert FFN: 32 tokens x (768 -> 3072 -> 768), relu, prob-weighted residual.
__global__ __launch_bounds__(256) void moe_ff_kernel(const unsigned short* __restrict__ pwi,
                                                     const unsigned short* __restrict__ pwo,
                                                     const unsigned short* __restrict__ xn,
                                                     const float* __restrict__ probs,
                                                     const int* __restrict__ bucket,
                                                     const int* __restrict__ meta,
                                                     const float* __restrict__ hidden,
                                                     float* __restrict__ out) {
  __shared__ __align__(16) unsigned short xs[MT * XROW];   // 48.5 KB normalized tokens (bf16)
  __shared__ __align__(16) unsigned short hsh[MT * HROW];  // 4.5 KB relu(h) chunk (bf16)
  __shared__ int   toks[MT];
  __shared__ float prb[MT];

  const int* counts = meta;
  const int* off    = meta + 16;
  const int* toff   = meta + 25;

  int b = blockIdx.x;
  if (b >= toff[8]) return;              // uniform early exit (before any barrier)
  int e = 0;
  while (e < 7 && b >= toff[e + 1]) ++e;
  int tile = b - toff[e];
  int row0 = off[e] + tile * MT;
  int rem  = counts[e] - tile * MT;

  int tid = threadIdx.x;
  if (tid < MT) {
    if (tid < rem) { int t = bucket[row0 + tid]; toks[tid] = t; prb[tid] = probs[t]; }
    else           { toks[tid] = -1; prb[tid] = 0.f; }
  }
  __syncthreads();

  // gather x rows into LDS (uint4 = 8 bf16 per transfer)
  for (int idx = tid; idx < MT * (kD / 8); idx += 256) {
    int r = idx / (kD / 8), c = idx % (kD / 8);
    int t = toks[r];
    uint4 val = make_uint4(0, 0, 0, 0);
    if (t >= 0) val = ((const uint4*)(xn + (long)t * kD))[c];
    *((uint4*)(xs + r * XROW + c * 8)) = val;
  }
  __syncthreads();

  int wid = tid >> 5, lane = tid & 31;
  int mt = wid & 1;                  // m-tile (0/1)
  int ng = wid >> 1;                 // n-group (0..3)
  int half = lane >> 4;              // A-layout lane half
  int mrow = mt * 16 + (lane & 15);  // A row this lane owns
  int aoff = half * 16;

  floatx8 yacc[12];
#pragma unroll
  for (int j = 0; j < 12; ++j) yacc[j] = (floatx8){0.f, 0.f, 0.f, 0.f, 0.f, 0.f, 0.f, 0.f};

  const unsigned short* wiBase = pwi + ((long)(e * kF16) * kD32) * 512;
  const unsigned short* woBase = pwo + ((long)(e * kD16) * kF32) * 512;

  for (int fc = 0; fc < NCH; ++fc) {
    // ---- FF1: h = relu(x @ wi[:, fc*64 + ng*16 .. +16]) ----
    floatx8 hacc = (floatx8){0.f, 0.f, 0.f, 0.f, 0.f, 0.f, 0.f, 0.f};
    const unsigned short* wiT = wiBase + ((long)(fc * 4 + ng) * kD32) * 512 + lane * 16;
    if (fc + 1 < NCH) __builtin_prefetch(wiT + (long)4 * kD32 * 512, 0, 0);
#pragma unroll 4
    for (int kt = 0; kt < kD32; ++kt) {
      bf16x16 a  = ldsFrag(xs + mrow * XROW + kt * 32 + aoff,
                           xs + mrow * XROW + kt * 32 + 8 + aoff);
      bf16x16 bb = gFrag(wiT + kt * 512);
      hacc = __builtin_amdgcn_wmma_f32_16x16x32_bf16(false, a, false, bb, (short)0, hacc,
                                                     false, false);
    }
    {
      int m0  = mt * 16 + half * 8;
      int col = ng * 16 + (lane & 15);
#pragma unroll
      for (int r = 0; r < 8; ++r) {
        float v = hacc[r];
        v = v > 0.f ? v : 0.f;                       // relu
        hsh[(m0 + r) * HROW + col] = f2bf(v);
      }
    }
    __syncthreads();

    // ---- FF2: y += h @ wo[fc*64 .. , :] ----
#pragma unroll
    for (int kk = 0; kk < 2; ++kk) {
      bf16x16 a2 = ldsFrag(hsh + mrow * HROW + kk * 32 + aoff,
                           hsh + mrow * HROW + kk * 32 + 8 + aoff);
#pragma unroll
      for (int j = 0; j < 12; ++j) {
        int nt = ng * 12 + j;
        const unsigned short* woT =
            woBase + ((long)nt * kF32 + (fc * 2 + kk)) * 512 + lane * 16;
        bf16x16 bb = gFrag(woT);
        yacc[j] = __builtin_amdgcn_wmma_f32_16x16x32_bf16(false, a2, false, bb, (short)0,
                                                          yacc[j], false, false);
      }
    }
    __syncthreads();   // hsh reused next chunk
  }

  // ---- epilogue: out = hidden + prob * y ----
#pragma unroll
  for (int r = 0; r < 8; ++r) {
    int m = mt * 16 + half * 8 + r;
    int t = toks[m];
    if (t < 0) continue;
    float p = prb[m];
    long base = (long)t * kD + (lane & 15);
#pragma unroll
    for (int j = 0; j < 12; ++j) {
      int d = (ng * 12 + j) * 16;
      out[base + d] = hidden[base + d] + p * yacc[j][r];
    }
  }
}

extern "C" void kernel_launch(void* const* d_in, const int* in_sizes, int n_in,
                              void* d_out, int out_size, void* d_ws, size_t ws_size,
                              hipStream_t stream) {
  const float* hidden = (const float*)d_in[0];
  const float* lnw    = (const float*)d_in[1];
  const float* rw     = (const float*)d_in[2];
  const float* wi     = (const float*)d_in[3];
  const float* wo     = (const float*)d_in[4];

  float* out        = (float*)d_out;
  float* logits_out = out + (long)kTok * kD;          // [B,S,E]
  float* idx_out    = logits_out + (long)kTok * kE;   // [B,S]

  char* ws = (char*)d_ws;
  unsigned short* pwi  = (unsigned short*)ws;                       // 37,748,736 B
  unsigned short* pwo  = (unsigned short*)(ws + 37748736);          // 37,748,736 B
  unsigned short* xn   = (unsigned short*)(ws + 75497472);          // 25,165,824 B
  float* probs         = (float*)(ws + 100663296);                  // 65,536 B
  int*   expert_of     = (int*)(ws + 100728832);                    // 65,536 B
  int*   bucket        = (int*)(ws + 100794368);                    // 65,536 B
  int*   meta          = (int*)(ws + 100859904);                    // 34 ints

  init_meta_kernel<<<1, 64, 0, stream>>>(meta);
  pack_kernel<<<(kE * kD * kF) / 256, 256, 0, stream>>>(wi, pwi, kD, kF);
  pack_kernel<<<(kE * kF * kD) / 256, 256, 0, stream>>>(wo, pwo, kF, kD);
  router_kernel<<<kTok / 8, 256, 0, stream>>>(hidden, lnw, rw, xn, probs, expert_of, meta,
                                              logits_out, idx_out);
  scan_kernel<<<1, 32, 0, stream>>>(meta);
  scatter_kernel<<<kTok / 256, 256, 0, stream>>>(expert_of, meta, bucket);
  moe_ff_kernel<<<kTok / MT + kE, 256, 0, stream>>>(pwi, pwo, xn, probs, bucket, meta,
                                                    hidden, out);
}
